// self_attention_49263274885453
// MI455X (gfx1250) — compile-verified, matrix operand order checked by reading, not run on hardware
//
#include <hip/hip_runtime.h>
#include <hip/hip_bf16.h>

typedef __attribute__((ext_vector_type(16))) __bf16 v16bf;
typedef __attribute__((ext_vector_type(8)))  float  v8f;
typedef unsigned short u16;

#define BB 16
#define SS 2048
#define DD 256
#define LSTR 2056   // f32 stride for LDS score rows (2048 + 8 pad)
#define ASTR 2072   // u16 stride for bf16 alpha rows (row starts conflict-free mod 64 banks)

// ---------- helpers ----------
__device__ inline u16 f2bf(float f) {
  union { __bf16 h; u16 u; } c;
  c.h = (__bf16)f;
  return c.u;
}

__device__ inline float wred_sum(float v) {
#pragma unroll
  for (int o = 16; o > 0; o >>= 1) v += __shfl_xor(v, o, 32);
  return v;
}
__device__ inline float wred_max(float v) {
#pragma unroll
  for (int o = 16; o > 0; o >>= 1) v = fmaxf(v, __shfl_xor(v, o, 32));
  return v;
}

// Load 16 bf16 elements of a 16x32 A / 32x16 B tile for one lane.
// `row` points at element [matrix_row, 0] with K contiguous.
// ISA 16-bit layout: elems 0..7  -> K = k0 + (hi?8:0) + 0..7
//                    elems 8..15 -> K = k0 + 16 + (hi?8:0) + 0..7
__device__ inline v16bf load_tile_bf16(const u16* row, int k0, bool hi) {
  const u16* p0 = row + k0 + (hi ? 8 : 0);
  v16bf r;
  u16* rp = (u16*)&r;
#pragma unroll
  for (int i = 0; i < 8; ++i) {
    rp[i]     = p0[i];
    rp[8 + i] = p0[16 + i];
  }
  return r;
}

// Same pattern from f32 source, converted to bf16 on the fly (used where the
// source is only ever read once, e.g. q projection input).
__device__ inline v16bf cvt_tile_f32(const float* row, int k0, bool hi) {
  const float* p0 = row + k0 + (hi ? 8 : 0);
  v16bf r;
#pragma unroll
  for (int i = 0; i < 8; ++i) {
    r[i]     = (__bf16)p0[i];
    r[8 + i] = (__bf16)p0[16 + i];
  }
  return r;
}

__device__ inline v8f wmma_bf16(v16bf a, v16bf b, v8f c) {
  return __builtin_amdgcn_wmma_f32_16x16x32_bf16(false, a, false, b, (short)0, c,
                                                 false, false);
}

// ---------- prep kernels ----------
__global__ void cvt_kv_kernel(const float* __restrict__ k, const float* __restrict__ v,
                              u16* __restrict__ kb, u16* __restrict__ vT) {
  int i = blockIdx.x * blockDim.x + threadIdx.x;
  if (i >= BB * SS * DD) return;
  kb[i] = f2bf(k[i]);
  int d = i % DD;
  int s = (i / DD) % SS;
  int b = i / (DD * SS);
  vT[((b * DD + d) * SS) + s] = f2bf(v[i]);   // V^T : [B, D, S], K-contiguous
}

__global__ void cvt_w_kernel(const float* __restrict__ Wq, const float* __restrict__ W1,
                             const float* __restrict__ W2, u16* __restrict__ WqT,
                             u16* __restrict__ W1T, u16* __restrict__ W2T) {
  int i = blockIdx.x * blockDim.x + threadIdx.x;  // i = n*256 + k
  int n = i >> 8, kk = i & 255;
  int src = kk * DD + n;
  WqT[i] = f2bf(Wq[src]);
  W1T[i] = f2bf(W1[src]);
  W2T[i] = f2bf(W2[src]);
}

// ---------- q projection: q_ = relu(q @ Wq + bq), bf16 out ----------
__global__ __launch_bounds__(256) void qproj_kernel(const float* __restrict__ q,
                                                    const u16* __restrict__ WqT,
                                                    const float* __restrict__ bq,
                                                    u16* __restrict__ qb) {
  int lane = threadIdx.x & 31, wave = threadIdx.x >> 5;
  bool hi = lane >= 16;
  int r16 = lane & 15;
  int row0 = blockIdx.x * 16;                       // row tile over B*S
  const float* arow = q + (long)(row0 + r16) * DD;

  v16bf A[8];
#pragma unroll
  for (int kk = 0; kk < 8; ++kk) A[kk] = cvt_tile_f32(arow, kk * 32, hi);

#pragma unroll
  for (int t = 0; t < 2; ++t) {
    int n0 = (wave * 2 + t) * 16;
    const u16* brow = WqT + (n0 + r16) * DD;
    v8f acc = {};
#pragma unroll
    for (int kk = 0; kk < 8; ++kk)
      acc = wmma_bf16(A[kk], load_tile_bf16(brow, kk * 32, hi), acc);
    int n = n0 + r16;
    float bias = bq[n];
#pragma unroll
    for (int r = 0; r < 8; ++r) {
      int m = r + (hi ? 8 : 0);
      float x = acc[r] + bias;
      x = x > 0.f ? x : 0.f;
      qb[(long)(row0 + m) * DD + n] = f2bf(x);
    }
  }
}

// ---------- entmax bisection for one register-resident row ----------
// alpha=2 => exponent 1/(alpha-1) = 1.
// Uses sum(max(x-t,0)) = sum(max(x,t)) - N*t in the 50-iter loop (2 VALU
// ops/elem, 4-way partials), exact final normalization.
__device__ inline void entmax_row(const float* __restrict__ x, int lane,
                                  float* __restrict__ gout, u16* __restrict__ lout) {
  float m0 = -3.4e38f, m1 = m0, m2 = m0, m3 = m0;
#pragma unroll
  for (int i = 0; i < 64; i += 4) {
    m0 = fmaxf(m0, x[i]);
    m1 = fmaxf(m1, x[i + 1]);
    m2 = fmaxf(m2, x[i + 2]);
    m3 = fmaxf(m3, x[i + 3]);
  }
  float mx = wred_max(fmaxf(fmaxf(m0, m1), fmaxf(m2, m3)));

  float tau = mx - 1.0f;                       // tau_lo
  float dm  = 1.0f - (1.0f / 2048.0f);         // tau_hi - tau_lo

  // exact f_lo (one pass; its sign gates the bisection updates)
  float s0 = 0.f, s1 = 0.f, s2 = 0.f, s3 = 0.f;
#pragma unroll
  for (int i = 0; i < 64; i += 4) {
    s0 += fmaxf(x[i] - tau, 0.f);
    s1 += fmaxf(x[i + 1] - tau, 0.f);
    s2 += fmaxf(x[i + 2] - tau, 0.f);
    s3 += fmaxf(x[i + 3] - tau, 0.f);
  }
  float f_lo = wred_sum((s0 + s1) + (s2 + s3)) - 1.0f;

  float tau_m = tau, fm = 0.f;
  for (int it = 0; it < 50; ++it) {
    dm *= 0.5f;
    tau_m = tau + dm;
    float t0 = 0.f, t1 = 0.f, t2 = 0.f, t3 = 0.f;
#pragma unroll
    for (int i = 0; i < 64; i += 4) {
      t0 += fmaxf(x[i], tau_m);
      t1 += fmaxf(x[i + 1], tau_m);
      t2 += fmaxf(x[i + 2], tau_m);
      t3 += fmaxf(x[i + 3], tau_m);
    }
    fm = wred_sum((t0 + t1) + (t2 + t3)) - 2048.0f * tau_m - 1.0f;
    tau = (fm * f_lo >= 0.f) ? tau_m : tau;
  }

  // exact normalizer at the final tau_m (ensure_sum_one)
  float e0 = 0.f, e1 = 0.f, e2 = 0.f, e3 = 0.f;
#pragma unroll
  for (int i = 0; i < 64; i += 4) {
    e0 += fmaxf(x[i] - tau_m, 0.f);
    e1 += fmaxf(x[i + 1] - tau_m, 0.f);
    e2 += fmaxf(x[i + 2] - tau_m, 0.f);
    e3 += fmaxf(x[i + 3] - tau_m, 0.f);
  }
  float inv = 1.0f / wred_sum((e0 + e1) + (e2 + e3));

#pragma unroll
  for (int i = 0; i < 64; ++i) {
    float p = fmaxf(x[i] - tau_m, 0.f) * inv;
    int c = lane + 32 * i;
    gout[c] = p;            // coalesced f32 stream to HBM
    lout[c] = f2bf(p);      // bf16 alpha row in LDS for the alpha@V WMMA pass
  }
}

// ---------- fused scores + entmax bisect + alpha@V ----------
__global__ __launch_bounds__(256) void attn_kernel(const u16* __restrict__ qb,
                                                   const u16* __restrict__ kb,
                                                   const u16* __restrict__ vT,
                                                   const unsigned char* __restrict__ mask,
                                                   float* __restrict__ attv,
                                                   float* __restrict__ alpha_out) {
  extern __shared__ float sc[];                 // [16][LSTR] fp32 scores
  u16* scb = (u16*)sc;                          // aliased bf16 alpha view [16][ASTR]
  int b  = blockIdx.x >> 7;                     // 128 query tiles per batch
  int m0 = (blockIdx.x & 127) * 16;
  int lane = threadIdx.x & 31, wave = threadIdx.x >> 5;
  bool hi = lane >= 16;
  int r16 = lane & 15;

  // A operand (q_ tile) resident in registers for the whole scores pass
  const u16* arow = qb + ((long)(b * SS + m0 + r16)) * DD;
  v16bf A[8];
#pragma unroll
  for (int kk = 0; kk < 8; ++kk) A[kk] = load_tile_bf16(arow, kk * 32, hi);

  const unsigned char* mrow = mask + b * SS;
  for (int nt = wave; nt < 128; nt += 8) {
    int n0 = nt * 16;
    const u16* brow = kb + ((long)(b * SS + n0 + r16)) * DD;
    v8f acc = {};
#pragma unroll
    for (int kk = 0; kk < 8; ++kk)
      acc = wmma_bf16(A[kk], load_tile_bf16(brow, kk * 32, hi), acc);
    int n = n0 + r16;
    bool msk = mrow[n] != 0;
#pragma unroll
    for (int r = 0; r < 8; ++r) {
      int m = r + (hi ? 8 : 0);
      float s = acc[r] * 0.0625f;               // 1/sqrt(256)
      sc[m * LSTR + n] = msk ? -1e15f : s;
    }
  }
  __syncthreads();

  // Pull BOTH of this wave's rows into registers before any wave starts
  // overwriting the buffer with bf16 alpha (barrier makes the aliasing safe).
  int mA = wave * 2, mB = wave * 2 + 1;
  float xA[64], xB[64];
#pragma unroll
  for (int i = 0; i < 64; ++i) {
    xA[i] = sc[mA * LSTR + lane + 32 * i];
    xB[i] = sc[mB * LSTR + lane + 32 * i];
  }
  __syncthreads();

  entmax_row(xA, lane, alpha_out + ((long)(b * SS + m0 + mA)) * SS, scb + mA * ASTR);
  entmax_row(xB, lane, alpha_out + ((long)(b * SS + m0 + mB)) * SS, scb + mB * ASTR);
  __syncthreads();

  // att_v = alpha @ V : K = 2048 in 64 WMMA steps; A direct bf16 from LDS,
  // B from V^T (K-contiguous bf16 in global, L2-resident per batch).
#pragma unroll 1
  for (int t = 0; t < 2; ++t) {
    int n0 = (wave * 2 + t) * 16;
    const u16* arow2 = scb + r16 * ASTR;
    const u16* brow = vT + ((long)(b * DD + n0 + r16)) * SS;
    v8f acc = {};
    for (int kk = 0; kk < 64; ++kk) {
      v16bf At = load_tile_bf16(arow2, kk * 32, hi);
      v16bf Bt = load_tile_bf16(brow, kk * 32, hi);
      acc = wmma_bf16(At, Bt, acc);
    }
    int n = n0 + r16;
#pragma unroll
    for (int r = 0; r < 8; ++r) {
      int m = r + (hi ? 8 : 0);
      attv[((long)(b * SS + m0 + m)) * DD + n] = acc[r];
    }
  }
}

// ---------- FFN + residual + LayerNorm ----------
__global__ __launch_bounds__(256) void ffn_kernel(const float* __restrict__ attv,
                                                  const u16* __restrict__ W1T,
                                                  const u16* __restrict__ W2T,
                                                  const float* __restrict__ b1,
                                                  const float* __restrict__ b2,
                                                  const float* __restrict__ gamma,
                                                  const float* __restrict__ beta,
                                                  float* __restrict__ out) {
  __shared__ float xt[16][DD + 8];    // f32 x (residual + final y buffer)
  __shared__ u16  xtb[16][DD + 8];    // bf16 x for GEMM1 A-operand
  __shared__ u16  htb[16][DD + 8];    // bf16 h for GEMM2 A-operand
  int row0 = blockIdx.x * 16;
  int lane = threadIdx.x & 31, wave = threadIdx.x >> 5;
  bool hi = lane >= 16;
  int r16 = lane & 15;

  for (int i = threadIdx.x; i < 16 * DD; i += 256) {
    int r = i >> 8, c = i & 255;
    float xv = attv[(long)(row0 + r) * DD + c];
    xt[r][c] = xv;
    xtb[r][c] = f2bf(xv);
  }
  __syncthreads();

  // h = relu(x @ W1 + b1) -> bf16 tile
#pragma unroll
  for (int t = 0; t < 2; ++t) {
    int n0 = (wave * 2 + t) * 16;
    const u16* brow = W1T + (n0 + r16) * DD;
    v8f acc = {};
#pragma unroll
    for (int kk = 0; kk < 8; ++kk)
      acc = wmma_bf16(load_tile_bf16(&xtb[r16][0], kk * 32, hi),
                      load_tile_bf16(brow, kk * 32, hi), acc);
    int n = n0 + r16;
    float bias = b1[n];
#pragma unroll
    for (int r = 0; r < 8; ++r) {
      int m = r + (hi ? 8 : 0);
      float x = acc[r] + bias;
      htb[m][n] = f2bf(x > 0.f ? x : 0.f);
    }
  }
  __syncthreads();

  // y = h @ W2 + b2 + x (held in regs; xt reused as y buffer after barrier)
  v8f y2[2];
#pragma unroll
  for (int t = 0; t < 2; ++t) {
    int n0 = (wave * 2 + t) * 16;
    const u16* brow = W2T + (n0 + r16) * DD;
    v8f acc = {};
#pragma unroll
    for (int kk = 0; kk < 8; ++kk)
      acc = wmma_bf16(load_tile_bf16(&htb[r16][0], kk * 32, hi),
                      load_tile_bf16(brow, kk * 32, hi), acc);
    int n = n0 + r16;
    float bias = b2[n];
#pragma unroll
    for (int r = 0; r < 8; ++r) {
      int m = r + (hi ? 8 : 0);
      y2[t][r] = acc[r] + bias + xt[m][n];
    }
  }
  __syncthreads();
#pragma unroll
  for (int t = 0; t < 2; ++t) {
    int n = (wave * 2 + t) * 16 + r16;
#pragma unroll
    for (int r = 0; r < 8; ++r) xt[r + (hi ? 8 : 0)][n] = y2[t][r];
  }
  __syncthreads();

  // LayerNorm per row (2 rows per wave, shuffle reductions)
#pragma unroll 1
  for (int rr = 0; rr < 2; ++rr) {
    int m = wave * 2 + rr;
    float yv[8], s = 0.f;
#pragma unroll
    for (int j = 0; j < 8; ++j) {
      yv[j] = xt[m][lane + 32 * j];
      s += yv[j];
    }
    float mu = wred_sum(s) * (1.0f / 256.0f);
    float v2 = 0.f;
#pragma unroll
    for (int j = 0; j < 8; ++j) {
      float d = yv[j] - mu;
      v2 += d * d;
    }
    float rs = rsqrtf(wred_sum(v2) * (1.0f / 256.0f) + 1e-5f);
#pragma unroll
    for (int j = 0; j < 8; ++j) {
      int c = lane + 32 * j;
      out[(long)(row0 + m) * DD + c] = (yv[j] - mu) * rs * gamma[c] + beta[c];
    }
  }
}

// ---------- launch ----------
extern "C" void kernel_launch(void* const* d_in, const int* in_sizes, int n_in,
                              void* d_out, int out_size, void* d_ws, size_t ws_size,
                              hipStream_t stream) {
  const float* q  = (const float*)d_in[0];
  const float* k  = (const float*)d_in[1];
  const float* v  = (const float*)d_in[2];
  const unsigned char* mask = (const unsigned char*)d_in[3];  // numpy bool, 1 byte
  const float* Wq = (const float*)d_in[4];
  const float* bq = (const float*)d_in[5];
  const float* W1 = (const float*)d_in[6];
  const float* b1 = (const float*)d_in[7];
  const float* W2 = (const float*)d_in[8];
  const float* b2 = (const float*)d_in[9];
  const float* gamma = (const float*)d_in[10];
  const float* beta  = (const float*)d_in[11];

  float* out_attv  = (float*)d_out;                       // [B,S,D]
  float* out_alpha = out_attv + (size_t)BB * SS * DD;     // [B,S,S]

  char* ws = (char*)d_ws;
  const size_t nqkv = (size_t)BB * SS * DD;
  u16* qb  = (u16*)ws;  ws += nqkv * 2;
  u16* kb  = (u16*)ws;  ws += nqkv * 2;
  u16* vT  = (u16*)ws;  ws += nqkv * 2;
  u16* WqT = (u16*)ws;  ws += (size_t)DD * DD * 2;
  u16* W1T = (u16*)ws;  ws += (size_t)DD * DD * 2;
  u16* W2T = (u16*)ws;  ws += (size_t)DD * DD * 2;
  float* attv = (float*)ws;                               // [B,S,D] fp32

  cvt_kv_kernel<<<(int)((nqkv + 255) / 256), 256, 0, stream>>>(k, v, kb, vT);
  cvt_w_kernel<<<(DD * DD) / 256, 256, 0, stream>>>(Wq, W1, W2, WqT, W1T, W2T);
  qproj_kernel<<<BB * SS / 16, 256, 0, stream>>>(q, WqT, bq, qb);

  size_t shmem = (size_t)16 * LSTR * sizeof(float);       // 128.5 KB of WGP LDS
  attn_kernel<<<BB * SS / 16, 256, shmem, stream>>>(qb, kb, vT, mask, attv, out_alpha);

  ffn_kernel<<<BB * SS / 16, 256, 0, stream>>>(attv, W1T, W2T, b1, b2, gamma, beta,
                                               out_attv);
}
